// Up_83674552861285
// MI455X (gfx1250) — matrix operand order repaired and checked by reading, not compiled, not run on hardware
//
#include <hip/hip_runtime.h>
#include <hip/hip_bf16.h>

// ---------------------------------------------------------------------------
// Sparse inverse-conv + BN/ReLU + concat-fuse for MI455X (gfx1250, wave32).
//
// Precision strategy: the whole pipeline is ~20 GFLOP vs ~0.66 GB of HBM
// traffic (~28us at 23.3 TB/s).  v_wmma_f32_16x16x32_bf16 with the 3-term
// bf16 hi/lo split (hi*hi + hi*lo + lo*hi, f32 accum) gives ~fp32 accuracy
// while keeping the kernels memory-bound; fp32 WMMA (16x16x4) would be ~8x
// slower on the matrix units and compute-bound.
// ---------------------------------------------------------------------------

typedef __attribute__((ext_vector_type(16))) __bf16 v16bf;
typedef __attribute__((ext_vector_type(8)))  float  v8f;

static constexpr int N_IN   = 100000;
static constexpr int N_OUT  = 400000;
static constexpr int KOFF   = 8;
static constexpr int PAIRS  = 100000;
static constexpr int C_IN   = 128;   // deconv input channels (also fuse K dim)
static constexpr int C_OUT  = 64;
static constexpr int TILES_A = 25;   // row-tiles per workgroup (deconv)
static constexpr int TILES_D = 25;   // row-tiles per workgroup (fuse)

__device__ __forceinline__ v8f wmma_bf16(v16bf a, v16bf b, v8f c) {
  // D = A(16x32 bf16) x B(32x16 bf16) + C(16x16 f32)
  return __builtin_amdgcn_wmma_f32_16x16x32_bf16(
      /*neg_a=*/false, a, /*neg_b=*/false, b,
      /*c_mod=*/(short)0, c, /*reuse_a=*/false, /*reuse_b=*/false);
}

// ------------------------- kernel 0: zero y + ws ---------------------------
__global__ __launch_bounds__(256) void zero_kernel(float* __restrict__ y,
                                                   float* __restrict__ ws,
                                                   long n4) {
  long i = (long)blockIdx.x * blockDim.x + threadIdx.x;
  long stride = (long)gridDim.x * blockDim.x;
  float4* y4 = (float4*)y;
  const float4 z = {0.f, 0.f, 0.f, 0.f};
  for (long j = i; j < n4; j += stride) y4[j] = z;
  if (blockIdx.x == 0 && threadIdx.x < 256) ws[threadIdx.x] = 0.f;
}

// -------------------- kernel A: gather-GEMM-scatter ------------------------
// grid.x = KOFF * (PAIRS/16/TILES_A); block = 128 threads (4 waves).
// Wave w owns output columns [16w, 16w+16).
__global__ __launch_bounds__(128) void deconv_scatter(
    const float* __restrict__ x, const float* __restrict__ Wd,
    const int* __restrict__ in_idx, const int* __restrict__ out_idx,
    float* __restrict__ y) {
  __shared__ __bf16 sAh[16 * C_IN];
  __shared__ __bf16 sAl[16 * C_IN];
  __shared__ int    sOid[16];

  const int tid  = threadIdx.x;
  const int wave = tid >> 5;
  const int lane = tid & 31;
  const int groupsPerK = (PAIRS / 16) / TILES_A;           // 250
  const int k     = blockIdx.x / groupsPerK;
  const int tile0 = (blockIdx.x % groupsPerK) * TILES_A;

  // --- B fragments: W_deconv[k] (128x64 f32) -> bf16 hi/lo, kept in VGPRs
  // and reused across all TILES_A row-tiles.  B layout (16-bit B 32x16,
  // this code's own contiguous k order, shared with A): lane holds column N=lane&15; lanes 0-15 carry K 0..15 of a
  // 32-step, lanes 16-31 carry K 16..31, packed as consecutive pairs.
  const float* Wk = Wd + (size_t)k * C_IN * C_OUT;
  const int ncol  = (wave << 4) + (lane & 15);
  const int kHalf = (lane >> 4) << 4;
  v16bf bh[4], bl[4];
#pragma unroll
  for (int s = 0; s < 4; ++s) {
#pragma unroll
    for (int e = 0; e < 16; ++e) {
      const int kk = s * 32 + kHalf + e;
      const float wv = Wk[kk * C_OUT + ncol];
      const __bf16 h = (__bf16)wv;
      bh[s][e] = h;
      bl[s][e] = (__bf16)(wv - (float)h);
    }
  }

  for (int t = 0; t < TILES_A; ++t) {
    const int p0 = (tile0 + t) * 16;

    // stage 16 gathered rows (bf16 hi/lo) into LDS: 8 threads per row,
    // 16 floats each (coalesced float4 from the gathered row).
    {
      const int r = tid >> 3, seg = tid & 7;
      const int irow = in_idx[(size_t)k * PAIRS + p0 + r];
      const float* src = x + (size_t)irow * C_IN + seg * 16;
#pragma unroll
      for (int j = 0; j < 16; j += 4) {
        const float4 v4 = *(const float4*)(src + j);
        const float vv[4] = {v4.x, v4.y, v4.z, v4.w};
#pragma unroll
        for (int q = 0; q < 4; ++q) {
          const float v = vv[q];
          const __bf16 h = (__bf16)v;
          sAh[r * C_IN + seg * 16 + j + q] = h;
          sAl[r * C_IN + seg * 16 + j + q] = (__bf16)(v - (float)h);
        }
      }
      if (tid < 16) sOid[tid] = out_idx[(size_t)k * PAIRS + p0 + tid];
    }
    __syncthreads();

    // 4 k-steps of 32, 3 split-WMMAs each, f32 accumulate.
    v8f c = {};
    const int m = lane & 15;
#pragma unroll
    for (int s = 0; s < 4; ++s) {
      v16bf ah, al;
#pragma unroll
      for (int e = 0; e < 16; ++e) {
        // A layout (matches B): lane<16 -> K 0..15; lane>=16 -> K 16..31.
        const int kk = s * 32 + ((lane >> 4) << 4) + e;
        ah[e] = sAh[m * C_IN + kk];
        al[e] = sAl[m * C_IN + kk];
      }
      c = wmma_bf16(ah, bh[s], c);   // hi*hi
      c = wmma_bf16(ah, bl[s], c);   // hi*lo
      c = wmma_bf16(al, bh[s], c);   // lo*hi  (lo*lo ~2^-32, dropped)
    }

    // scatter-accumulate the 16x16 D tile.  C/D layout: VGPR r holds
    // M=r (lanes 0-15) / M=8+r (lanes 16-31), N=lane&15.
    const int mbase = (lane >> 4) << 3;
#pragma unroll
    for (int r = 0; r < 8; ++r) {
      const int orow = sOid[mbase + r];
      atomicAdd(&y[(size_t)orow * C_OUT + ncol], c[r]);
    }
    __syncthreads();
  }
}

// ---------------- kernel B: per-channel sum / sum-of-squares ---------------
__global__ __launch_bounds__(256) void bn_stats(const float* __restrict__ y,
                                                float* __restrict__ ws,
                                                int nrows) {
  const int c   = threadIdx.x & 63;
  const int sub = threadIdx.x >> 6;   // 4 row-lanes per channel
  float s = 0.f, s2 = 0.f;
  for (int n = blockIdx.x * 4 + sub; n < nrows; n += gridDim.x * 4) {
    const float v = y[(size_t)n * C_OUT + c];
    s += v; s2 += v * v;
  }
  __shared__ float red[256], red2[256];
  red[threadIdx.x] = s; red2[threadIdx.x] = s2;
  __syncthreads();
  if (sub == 0) {
    s  = red[c]  + red[64 + c]  + red[128 + c]  + red[192 + c];
    s2 = red2[c] + red2[64 + c] + red2[128 + c] + red2[192 + c];
    atomicAdd(&ws[c], s);
    atomicAdd(&ws[64 + c], s2);
  }
}

// -------------------- kernel C: finalize BN scale/shift --------------------
__global__ void bn_final(const float* __restrict__ sums,
                         const float* __restrict__ gamma,
                         const float* __restrict__ beta,
                         float* __restrict__ ss, int nrows) {
  const int c = threadIdx.x;
  if (c < C_OUT) {
    const float inv_n = 1.0f / (float)nrows;
    const float mean  = sums[c] * inv_n;
    const float var   = sums[64 + c] * inv_n - mean * mean;
    const float scl   = gamma[c] * rsqrtf(var + 1e-5f);
    ss[c]       = scl;
    ss[64 + c]  = beta[c] - mean * scl;
  }
}

// --------- kernel D: [relu(bn(y)) | skip] @ W_fuse, in-place over y --------
// grid.x = (N_OUT/16)/TILES_D; block = 128 threads (4 waves).
__global__ __launch_bounds__(128) void fuse_gemm(
    const float* __restrict__ yin, const float* __restrict__ skip,
    const float* __restrict__ Wf, const float* __restrict__ ss,
    float* __restrict__ out) {
  __shared__ __bf16 sAh[16 * C_IN];
  __shared__ __bf16 sAl[16 * C_IN];
  __shared__ float  sScale[64], sShift[64];

  const int tid  = threadIdx.x;
  const int wave = tid >> 5;
  const int lane = tid & 31;
  if (tid < 64) { sScale[tid] = ss[tid]; sShift[tid] = ss[64 + tid]; }

  // B fragments from W_fuse (128x64), same layout as kernel A.
  const int ncol  = (wave << 4) + (lane & 15);
  const int kHalf = (lane >> 4) << 4;
  v16bf bh[4], bl[4];
#pragma unroll
  for (int s = 0; s < 4; ++s) {
#pragma unroll
    for (int e = 0; e < 16; ++e) {
      const int kk = s * 32 + kHalf + e;
      const float wv = Wf[kk * C_OUT + ncol];
      const __bf16 h = (__bf16)wv;
      bh[s][e] = h;
      bl[s][e] = (__bf16)(wv - (float)h);
    }
  }
  __syncthreads();

  const int tile0 = blockIdx.x * TILES_D;
  for (int t = 0; t < TILES_D; ++t) {
    const int n0 = (tile0 + t) * 16;
    // stage A rows: cols 0..63 = relu(bn(y)); cols 64..127 = skip.
    {
      const int r = tid >> 3, seg = tid & 7;
      const int row = n0 + r;
      const int cb  = seg * 16;
#pragma unroll
      for (int j = 0; j < 16; ++j) {
        const int col = cb + j;
        float v;
        if (col < C_OUT) {
          v = yin[(size_t)row * C_OUT + col];
          v = fmaxf(v * sScale[col] + sShift[col], 0.0f);
        } else {
          v = skip[(size_t)row * C_OUT + (col - C_OUT)];
        }
        const __bf16 h = (__bf16)v;
        sAh[r * C_IN + col] = h;
        sAl[r * C_IN + col] = (__bf16)(v - (float)h);
      }
    }
    __syncthreads();

    v8f c = {};
    const int m = lane & 15;
#pragma unroll
    for (int s = 0; s < 4; ++s) {
      v16bf ah, al;
#pragma unroll
      for (int e = 0; e < 16; ++e) {
        const int kk = s * 32 + ((lane >> 4) << 4) + e;
        ah[e] = sAh[m * C_IN + kk];
        al[e] = sAl[m * C_IN + kk];
      }
      c = wmma_bf16(ah, bh[s], c);
      c = wmma_bf16(ah, bl[s], c);
      c = wmma_bf16(al, bh[s], c);
    }

    const int mbase = (lane >> 4) << 3;
#pragma unroll
    for (int r = 0; r < 8; ++r)
      out[(size_t)(n0 + mbase + r) * C_OUT + ncol] = c[r];
    __syncthreads();
  }
}

// ---------------------------------------------------------------------------
extern "C" void kernel_launch(void* const* d_in, const int* in_sizes, int n_in,
                              void* d_out, int out_size, void* d_ws, size_t ws_size,
                              hipStream_t stream) {
  const float* x      = (const float*)d_in[0];   // [N_IN, 128]
  const float* skip   = (const float*)d_in[1];   // [N_OUT, 64]
  const float* Wd     = (const float*)d_in[2];   // [8, 128, 64]
  const float* gamma  = (const float*)d_in[3];   // [64]
  const float* beta   = (const float*)d_in[4];   // [64]
  const float* Wf     = (const float*)d_in[5];   // [128, 64]
  const int*   in_idx = (const int*)d_in[6];     // [8, P]
  const int*   out_idx= (const int*)d_in[7];     // [8, P]
  float* out = (float*)d_out;                    // [N_OUT, 64]; doubles as y
  float* ws  = (float*)d_ws;                     // 256 floats: sum|sumsq|scale|shift

  (void)in_sizes; (void)n_in; (void)out_size; (void)ws_size;

  const long n4 = (long)N_OUT * C_OUT / 4;
  zero_kernel<<<2048, 256, 0, stream>>>(out, ws, n4);

  const int gridA = KOFF * ((PAIRS / 16) / TILES_A);        // 2000
  deconv_scatter<<<gridA, 128, 0, stream>>>(x, Wd, in_idx, out_idx, out);

  bn_stats<<<512, 256, 0, stream>>>(out, ws, N_OUT);
  bn_final<<<1, 64, 0, stream>>>(ws, gamma, beta, ws + 128, N_OUT);

  const int gridD = (N_OUT / 16) / TILES_D;                 // 1000
  fuse_gemm<<<gridD, 128, 0, stream>>>(out, skip, Wf, ws + 128, out);
}
